// HGT_DNF_57793079935293
// MI455X (gfx1250) — matrix operand order repaired and checked by reading, not compiled, not run on hardware
//
#include <hip/hip_runtime.h>
#include <hip/hip_bf16.h>

// CDNA5 / gfx1250 wave32 WMMA + packed-f16 types
typedef __attribute__((ext_vector_type(16))) _Float16 v16h;
typedef __attribute__((ext_vector_type(8)))  _Float16 v8h;
typedef __attribute__((ext_vector_type(4)))  _Float16 v4h;
typedef __attribute__((ext_vector_type(2)))  _Float16 v2h;
typedef __attribute__((ext_vector_type(8)))  float    v8f;
typedef __attribute__((ext_vector_type(2)))  float    vf2;

#define NPRED  106          // K (predicates)
#define KP     128          // K zero-padded extent in LDS (covers all 4 WMMA windows)
#define STR    136          // LDS row stride in halves (16B-aligned, bank-spread)
#define ODIM   1024         // O (clauses)
#define BM     32           // macro-tile rows (batch)  -> 2 wave-rows of 16
#define BN     64           // macro-tile cols (out)    -> 4 wave-cols of 16
#define DELTA_ 0.01f

static __device__ inline float hmax8(v8h v) {
    v4h a = __builtin_shufflevector(v, v, 0, 1, 2, 3);
    v4h b = __builtin_shufflevector(v, v, 4, 5, 6, 7);
    v4h m = __builtin_elementwise_max(a, b);
    v2h c = __builtin_shufflevector(m, m, 0, 1);
    v2h d = __builtin_shufflevector(m, m, 2, 3);
    v2h e = __builtin_elementwise_max(c, d);
    _Float16 f = (e[0] > e[1]) ? e[0] : e[1];
    return (float)f;
}

__global__ __launch_bounds__(256, 2)
void hgt_dnf_wmma(const float* __restrict__ x,     // [B][106]
                  const float* __restrict__ w,     // [54][1024]
                  const int*   __restrict__ idx,   // [106]
                  float*       __restrict__ out,   // [B][1024]
                  int Bsz)
{
    // Four pre-converted f16 tiles, fragment-friendly layout [row][k]:
    //   xma: masked(x)  [BM][STR]  (GEMM-A main)
    //   xab: |x|        [BM][STR]  (GEMM-A abs + max loop)
    //   wsg: W^T        [BN][STR]  (GEMM-B main)
    //   wab: |W|^T      [BN][STR]  (GEMM-B abs + max loop)
    __shared__ __align__(16) _Float16 smem[(BM + BM + BN + BN) * STR];
    __shared__ int sidx[NPRED];
    _Float16* const xma = smem;
    _Float16* const xab = smem + BM * STR;
    _Float16* const wsg = smem + 2 * BM * STR;
    _Float16* const wab = smem + 2 * BM * STR + BN * STR;

    const int tid = threadIdx.x;
    const int b0  = blockIdx.y * BM;
    const int o0  = blockIdx.x * BN;

    // ---- stage idx into LDS (broadcast source for the W gather) ----
    if (tid < NPRED) sidx[tid] = idx[tid];

    // ---- stage x tile: one b64 global load -> {masked f16, |.| f16} ----
    // Thread handles (row m, dword-col j) => halves k=2j, 2j+1. NPRED is even
    // and row bases are 8B-aligned (106*4 % 8 == 0), so float2 loads are legal
    // and pairs never straddle the K boundary.
    for (int i = tid; i < BM * (KP / 2); i += 256) {
        const int m = i >> 6, j = i & 63;       // KP/2 == 64
        const int gm = b0 + m;
        const int k0 = 2 * j;
        vf2 p = {0.0f, 0.0f};
        if (k0 < NPRED && gm < Bsz) p = *(const vf2*)(x + gm * NPRED + k0);
        v2h pm, pa;
        pm[0] = (_Float16)((p[0] >= -1.0f) ? p[0] : 0.0f);
        pm[1] = (_Float16)((p[1] >= -1.0f) ? p[1] : 0.0f);
        pa[0] = (_Float16)fabsf(p[0]);
        pa[1] = (_Float16)fabsf(p[1]);
        *(v2h*)&xma[m * STR + k0] = pm;
        *(v2h*)&xab[m * STR + k0] = pa;
    }
    __syncthreads();   // sidx (and x tile ordering) ready

    // ---- stage W tile transposed via sidx gather (o spans lanes: coalesced) ----
    for (int i = tid; i < BN * (KP / 2); i += 256) {
        const int o = i & 63, j = i >> 6;
        const int k0 = 2 * j, k1 = 2 * j + 1;
        float a = (k0 < NPRED) ? w[sidx[k0] * ODIM + (o0 + o)] : 0.0f;
        float b = (k1 < NPRED) ? w[sidx[k1] * ODIM + (o0 + o)] : 0.0f;
        v2h ps, pa;
        ps[0] = (_Float16)a;
        ps[1] = (_Float16)b;
        pa[0] = (_Float16)fabsf(a);
        pa[1] = (_Float16)fabsf(b);
        *(v2h*)&wsg[o * STR + k0] = ps;
        *(v2h*)&wab[o * STR + k0] = pa;
    }
    __syncthreads();

    const int warp = tid >> 5;
    const int lane = tid & 31;
    const int wm   = (warp >> 2) << 4;   // 0 / 16
    const int wn   = (warp &  3) << 4;   // 0 / 16 / 32 / 48
    const int col  = lane & 15;
    const int hif  = lane >> 4;

    v8f acc_m = {};   // masked-x GEMM accumulator
    v8f acc_a = {};   // |x|*|W| GEMM accumulator (sum term)

    const _Float16* __restrict__ amrow = &xma[(wm + col) * STR];  // A: M = lane%16
    const _Float16* __restrict__ aarow = &xab[(wm + col) * STR];
    const _Float16* __restrict__ bsrow = &wsg[(wn + col) * STR];  // B: N = lane%16
    const _Float16* __restrict__ barow = &wab[(wn + col) * STR];

    // ---- 2 GEMMs over K=128: fragments are pure ds_load_b128 + shuffle ----
#pragma unroll
    for (int kw = 0; kw < 4; ++kw) {
        const int k0 = kw * 32;
        // A 16x32 f16 layout: halves 0..7  at K = k0 + 8*hif + h
        //                     halves 8..15 at K = k0 + 16 + 8*hif + (h-8)
        const int ka0 = k0 + (hif << 3);
        v8h am_lo = *(const v8h*)(amrow + ka0);
        v8h am_hi = *(const v8h*)(amrow + ka0 + 16);
        v8h aa_lo = *(const v8h*)(aarow + ka0);
        v8h aa_hi = *(const v8h*)(aarow + ka0 + 16);
        // B 32x16 f16 layout: same K mapping as A (8*hif + h, +16 for halves 8..15)
        const int kb0 = k0 + (hif << 3);
        v8h bs_lo = *(const v8h*)(bsrow + kb0);
        v8h bs_hi = *(const v8h*)(bsrow + kb0 + 16);
        v8h ba_lo = *(const v8h*)(barow + kb0);
        v8h ba_hi = *(const v8h*)(barow + kb0 + 16);

        v16h a_m = __builtin_shufflevector(am_lo, am_hi, 0,1,2,3,4,5,6,7,8,9,10,11,12,13,14,15);
        v16h a_a = __builtin_shufflevector(aa_lo, aa_hi, 0,1,2,3,4,5,6,7,8,9,10,11,12,13,14,15);
        v16h b_s = __builtin_shufflevector(bs_lo, bs_hi, 0,1,2,3,4,5,6,7,8,9,10,11,12,13,14,15);
        v16h b_a = __builtin_shufflevector(ba_lo, ba_hi, 0,1,2,3,4,5,6,7,8,9,10,11,12,13,14,15);

        acc_m = __builtin_amdgcn_wmma_f32_16x16x32_f16(
                    false, a_m, false, b_s, (short)0, acc_m, false, false);
        acc_a = __builtin_amdgcn_wmma_f32_16x16x32_f16(
                    false, a_a, false, b_a, (short)0, acc_a, false, false);
    }

    // ---- max_n |x[b,n]|*|W[n,o]| in C-fragment layout, packed f16:
    //      per 8 K-elements: 9x ds_load_b128 + v_pk_mul_f16/v_pk_max chains.
    //      All terms >= 0 and the LDS pad is 0, so zero-init and a padded
    //      extent (112 halves >= 106) are exact.
    v8h mxv[8];
#pragma unroll
    for (int r = 0; r < 8; ++r) mxv[r] = (v8h)(_Float16)0.0f;

#pragma unroll 1
    for (int n = 0; n < 112; n += 8) {
        const v8h w8 = *(const v8h*)(barow + n);
#pragma unroll
        for (int r = 0; r < 8; ++r) {
            const v8h x8 = *(const v8h*)&xab[(wm + r + (hif << 3)) * STR + n];
            mxv[r] = __builtin_elementwise_max(mxv[r], (v8h)(w8 * x8));
        }
    }

    // ---- epilogue: out = main + DELTA * (max - sum_abs) ----
    float bias[8];
#pragma unroll
    for (int r = 0; r < 8; ++r) bias[r] = hmax8(mxv[r]);

    const int mbase = b0 + wm + (hif << 3);
    const int o     = o0 + wn + col;
    if (b0 + BM <= Bsz) {
        // Full tile (always the case when B % 32 == 0): branchless stores.
#pragma unroll
        for (int r = 0; r < 8; ++r)
            out[(mbase + r) * ODIM + o] = acc_m[r] + DELTA_ * (bias[r] - acc_a[r]);
    } else {
#pragma unroll
        for (int r = 0; r < 8; ++r)
            if (mbase + r < Bsz)
                out[(mbase + r) * ODIM + o] = acc_m[r] + DELTA_ * (bias[r] - acc_a[r]);
    }
}

extern "C" void kernel_launch(void* const* d_in, const int* in_sizes, int n_in,
                              void* d_out, int out_size, void* d_ws, size_t ws_size,
                              hipStream_t stream) {
    const float* x   = (const float*)d_in[0];
    const float* w   = (const float*)d_in[1];
    const int*   idx = (const int*)d_in[2];
    float*       out = (float*)d_out;

    const int Bsz = in_sizes[0] / NPRED;          // 4096
    dim3 grid(ODIM / BN, (Bsz + BM - 1) / BM);    // (16, 128)
    dim3 block(256);
    hgt_dnf_wmma<<<grid, block, 0, stream>>>(x, w, idx, out, Bsz);
}
